// NodeBlock_69346541961223
// MI455X (gfx1250) — matrix operand order repaired and checked by reading, not compiled, not run on hardware
//
#include <hip/hip_runtime.h>
#include <hip/hip_bf16.h>

// ---------------- problem constants (match reference) ----------------
#define N_NODES 50000
#define N_EDGES 800000
#define D_EDGE  64
#define D_NODE  64
#define D_HID   256
#define D_OUT   64
#define N_TILES (N_NODES / 16)          // 3125
#define N_PAIRS ((N_TILES + 1) / 2)     // 1563 (last pair is a singleton)
#define WAVES_PER_BLOCK 4

typedef __attribute__((ext_vector_type(16))) __bf16 v16bf;
typedef __attribute__((ext_vector_type(8)))  float  v8f;

union ABFrag {
    unsigned int u[8];
    uint4        q[2];
    v16bf        v;
};

// f32 -> bf16 (round-to-nearest-even), bit version (used in small pack kernel)
__device__ __forceinline__ unsigned short f2bf(float f) {
    unsigned int x = __builtin_bit_cast(unsigned int, f);
    x += 0x7fffu + ((x >> 16) & 1u);
    return (unsigned short)(x >> 16);
}

// ---------------- kernel 1: zero the aggregate buffers ----------------
__global__ void zero_kernel(float4* p, int n4) {
    int i = blockIdx.x * blockDim.x + threadIdx.x;
    if (i < n4) p[i] = make_float4(0.f, 0.f, 0.f, 0.f);
}

// ---------------- kernel 2: pack W1/W2 into bf16 B-fragment layout ----------------
// B-fragment (16-bit B, KxN=32x16, wave32): lane = {n = lane&15, half = lane>>4},
// VGPR j holds K = (j>>2)*16 + half*8 + 2(j&3), +1 (within the 32-wide K window).
// Packed linear index: ((kb*NT + nt)*32 + lane)*8 + j  -> coalesced uint4 loads in GEMM.
__global__ void pack_weights_kernel(const float* __restrict__ W1,
                                    const float* __restrict__ W2,
                                    unsigned int* __restrict__ w1p,
                                    unsigned int* __restrict__ w2p) {
    int t = blockIdx.x * blockDim.x + threadIdx.x;
    if (t < 6 * 16 * 32 * 8) {                      // W1: [192,256] -> 6 kb x 16 nt
        int j    = t & 7;
        int lane = (t >> 3) & 31;
        int nt   = (t >> 8) & 15;
        int kb   = t >> 12;
        int n    = lane & 15;
        int half = lane >> 4;
        int k    = kb * 32 + half * 8 + (j >> 2) * 16 + 2 * (j & 3);
        int col  = nt * 16 + n;
        float lo = W1[(size_t)k * D_HID + col];
        float hi = W1[(size_t)(k + 1) * D_HID + col];
        w1p[t] = (unsigned int)f2bf(lo) | ((unsigned int)f2bf(hi) << 16);
    } else if (t < 6 * 16 * 32 * 8 + 8 * 4 * 32 * 8) {   // W2: [256,64] -> 8 kb x 4 nt
        int u    = t - 6 * 16 * 32 * 8;
        int j    = u & 7;
        int lane = (u >> 3) & 31;
        int nt   = (u >> 8) & 3;
        int kb   = u >> 10;
        int n    = lane & 15;
        int half = lane >> 4;
        int k    = kb * 32 + half * 8 + (j >> 2) * 16 + 2 * (j & 3);
        int col  = nt * 16 + n;
        float lo = W2[(size_t)k * D_OUT + col];
        float hi = W2[(size_t)(k + 1) * D_OUT + col];
        w2p[u] = (unsigned int)f2bf(lo) | ((unsigned int)f2bf(hi) << 16);
    }
}

// ---------------- kernel 3: edge scatter (segment sums) ----------------
__global__ void scatter_kernel(const float* __restrict__ edge_feat,
                               const int* __restrict__ senders,
                               const int* __restrict__ receivers,
                               float* __restrict__ in_agg,
                               float* __restrict__ out_agg) {
    unsigned int t = blockIdx.x * blockDim.x + threadIdx.x;
    unsigned int e = t >> 6;
    unsigned int d = t & 63;
    if (e >= N_EDGES) return;
    float v = edge_feat[(size_t)e * D_EDGE + d];
    int r = receivers[e];
    int s = senders[e];
    unsafeAtomicAdd(&in_agg[(size_t)r * D_EDGE + d], v);
    unsafeAtomicAdd(&out_agg[(size_t)s * D_EDGE + d], v);
}

// ---------------- helper: pack 8 consecutive-K floats into av[base..base+7] ----------------
__device__ __forceinline__ void cvt8(v16bf& av, int base, float4 a, float4 b) {
    av[base + 0] = (__bf16)a.x; av[base + 1] = (__bf16)a.y;
    av[base + 2] = (__bf16)a.z; av[base + 3] = (__bf16)a.w;
    av[base + 4] = (__bf16)b.x; av[base + 5] = (__bf16)b.y;
    av[base + 6] = (__bf16)b.z; av[base + 7] = (__bf16)b.w;
}

// load + convert one 16x32 bf16 A fragment; per-lane K = half*8 + 0..7, then 16 + half*8 + 0..7
__device__ __forceinline__ ABFrag load_a_frag(const float* rowp) {
    float4 f0 = *(const float4*)(rowp + 0);    // K = half*8 + 0..3
    float4 f1 = *(const float4*)(rowp + 4);    // K = half*8 + 4..7
    float4 f2 = *(const float4*)(rowp + 16);   // K = 16 + half*8 + 0..3
    float4 f3 = *(const float4*)(rowp + 20);   // K = 16 + half*8 + 4..7
    ABFrag a;
    v16bf av;
    cvt8(av, 0, f0, f1);
    cvt8(av, 8, f2, f3);
    a.v = av;
    return a;
}

// ---------------- kernel 4: fused MLP via bf16 WMMA ----------------
// One wave32 per PAIR of 16-node tiles (M=32 per wave): every B fragment feeds two
// independent WMMA chains, halving weight re-fetch traffic. The nt loops are kept as
// REAL loops (#pragma unroll 1) so only one iteration's B fragments + 2 accumulators
// are live -> ~180 VGPRs, no vgpr_msb addressing, good per-iteration load/WMMA overlap.
__global__ __launch_bounds__(32 * WAVES_PER_BLOCK)
void gemm_kernel(const float* __restrict__ in_agg,
                 const float* __restrict__ out_agg,
                 const float* __restrict__ node_feat,
                 const unsigned int* __restrict__ w1p,
                 const float* __restrict__ b1,
                 const unsigned int* __restrict__ w2p,
                 const float* __restrict__ b2,
                 float* __restrict__ out) {
    __shared__ unsigned short lds_h[WAVES_PER_BLOCK * 2 * 16 * D_HID];   // 64 KB

    const int wave = threadIdx.x >> 5;
    const int lane = threadIdx.x & 31;
    const int pair = blockIdx.x * WAVES_PER_BLOCK + wave;
    if (pair >= N_PAIRS) return;          // wave-uniform; EXEC all-ones for WMMA

    const int ln   = lane & 15;
    const int half = lane >> 4;

    const int  t0    = pair * 2;
    const bool has1  = (t0 + 1) < N_TILES;
    const int  t1    = has1 ? (t0 + 1) : t0;       // clamp loads; skip dup store later
    const int  base0 = t0 * 16;
    const int  base1 = t1 * 16;

    const float* srcs[3] = { in_agg, out_agg, node_feat };

    // ---- load + convert A fragments for both row-tiles (2 x 48 VGPRs) ----
    ABFrag afr0[6], afr1[6];
#pragma unroll
    for (int kb = 0; kb < 6; ++kb) {
        const float* s   = srcs[kb >> 1];
        const int    off = (kb & 1) * 32 + half * 8;
        afr0[kb] = load_a_frag(s + (size_t)(base0 + ln) * 64 + off);
        afr1[kb] = load_a_frag(s + (size_t)(base1 + ln) * 64 + off);
    }

    unsigned short* hbuf0 = lds_h + wave * (2 * 16 * D_HID);
    unsigned short* hbuf1 = hbuf0 + 16 * D_HID;

    // ---- GEMM1: h = relu(X @ W1 + b1), two independent chains per B fragment ----
#pragma unroll 1
    for (int nt = 0; nt < 16; ++nt) {
        ABFrag bfr[6];
#pragma unroll
        for (int kb = 0; kb < 6; ++kb) {
            const uint4* wp = (const uint4*)(w1p + ((size_t)(kb * 16 + nt) * 32 + lane) * 8);
            bfr[kb].q[0] = wp[0];
            bfr[kb].q[1] = wp[1];
        }
        const float bv = b1[nt * 16 + ln];
        v8f acc0, acc1;
#pragma unroll
        for (int r = 0; r < 8; ++r) { acc0[r] = bv; acc1[r] = bv; }
#pragma unroll
        for (int kb = 0; kb < 6; ++kb) {
            acc0 = __builtin_amdgcn_wmma_f32_16x16x32_bf16(
                false, afr0[kb].v, false, bfr[kb].v, (short)0, acc0, false, false);
            acc1 = __builtin_amdgcn_wmma_f32_16x16x32_bf16(
                false, afr1[kb].v, false, bfr[kb].v, (short)0, acc1, false, false);
        }
        // ReLU + stage (C/D layout row = r + 8*half, col = nt*16 + ln)
#pragma unroll
        for (int r = 0; r < 8; ++r) {
            float v0 = __builtin_fmaxf(acc0[r], 0.f);
            float v1 = __builtin_fmaxf(acc1[r], 0.f);
            const int idx = (r + 8 * half) * D_HID + nt * 16 + ln;
            hbuf0[idx] = __builtin_bit_cast(unsigned short, (__bf16)v0);
            hbuf1[idx] = __builtin_bit_cast(unsigned short, (__bf16)v1);
        }
    }

    // ---- GEMM2: out = h @ W2 + b2, row-tiles processed sequentially (reuse regs) ----
#pragma unroll
    for (int s = 0; s < 2; ++s) {
        const unsigned short* hb = (s == 0) ? hbuf0 : hbuf1;

        ABFrag a2[8];
#pragma unroll
        for (int kb = 0; kb < 8; ++kb) {
            const unsigned short* hp = hb + ln * D_HID + kb * 32 + half * 8;
            a2[kb].q[0] = *(const uint4*)(hp);        // K = k0 + half*8 + 0..7
            a2[kb].q[1] = *(const uint4*)(hp + 16);   // K = k0 + 16 + half*8 + 0..7
        }

        const bool do_store = (s == 0) || has1;       // wave-uniform
        const int  baseS    = (s == 0) ? base0 : base1;

#pragma unroll 1
        for (int nt = 0; nt < 4; ++nt) {
            ABFrag bfr[8];
#pragma unroll
            for (int kb = 0; kb < 8; ++kb) {
                const uint4* wp = (const uint4*)(w2p + ((size_t)(kb * 4 + nt) * 32 + lane) * 8);
                bfr[kb].q[0] = wp[0];
                bfr[kb].q[1] = wp[1];
            }
            const float bv = b2[nt * 16 + ln];
            v8f acc;
#pragma unroll
            for (int r = 0; r < 8; ++r) acc[r] = bv;
#pragma unroll
            for (int kb = 0; kb < 8; ++kb) {
                acc = __builtin_amdgcn_wmma_f32_16x16x32_bf16(
                    false, a2[kb].v, false, bfr[kb].v, (short)0, acc, false, false);
            }
            if (do_store) {
#pragma unroll
                for (int r = 0; r < 8; ++r) {
                    out[(size_t)(baseS + r + 8 * half) * D_OUT + nt * 16 + ln] = acc[r];
                }
            }
        }
    }
}

// ---------------- launcher ----------------
extern "C" void kernel_launch(void* const* d_in, const int* in_sizes, int n_in,
                              void* d_out, int out_size, void* d_ws, size_t ws_size,
                              hipStream_t stream) {
    const float* node_feat = (const float*)d_in[0];
    const float* edge_feat = (const float*)d_in[1];
    const int*   senders   = (const int*)d_in[2];
    const int*   receivers = (const int*)d_in[3];
    const float* W1        = (const float*)d_in[4];
    const float* b1        = (const float*)d_in[5];
    const float* W2        = (const float*)d_in[6];
    const float* b2        = (const float*)d_in[7];
    float* out = (float*)d_out;

    // workspace layout
    const size_t AGG = (size_t)N_NODES * D_EDGE;          // 3,200,000 floats
    float* in_agg  = (float*)d_ws;
    float* out_agg = in_agg + AGG;
    unsigned int* w1p = (unsigned int*)(out_agg + AGG);   // 24576 u32
    unsigned int* w2p = w1p + 6 * 16 * 32 * 8;            //  8192 u32

    // 1) zero aggregates
    {
        int n4 = (int)(2 * AGG / 4);
        zero_kernel<<<(n4 + 255) / 256, 256, 0, stream>>>((float4*)in_agg, n4);
    }
    // 2) pack weights to bf16 fragment order
    {
        int total = 6 * 16 * 32 * 8 + 8 * 4 * 32 * 8;     // 32768
        pack_weights_kernel<<<(total + 255) / 256, 256, 0, stream>>>(W1, W2, w1p, w2p);
    }
    // 3) edge scatter
    {
        unsigned int total = (unsigned int)N_EDGES * 64u;  // 51.2M threads
        scatter_kernel<<<(total + 255) / 256, 256, 0, stream>>>(
            edge_feat, senders, receivers, in_agg, out_agg);
    }
    // 4) fused MLP (one wave per pair of 16-row tiles)
    {
        int blocks = (N_PAIRS + WAVES_PER_BLOCK - 1) / WAVES_PER_BLOCK;  // 391
        gemm_kernel<<<blocks, 32 * WAVES_PER_BLOCK, 0, stream>>>(
            in_agg, out_agg, node_feat, w1p, b1, w2p, b2, out);
    }
}
